// AGConv_16922171146621
// MI455X (gfx1250) — matrix operand order repaired and checked by reading, not compiled, not run on hardware
//
#include <hip/hip_runtime.h>
#include <hip/hip_bf16.h>

typedef __attribute__((ext_vector_type(16))) _Float16 v16h;
typedef __attribute__((ext_vector_type(8)))  float    v8f;

#define BATCH 4
#define CIN   6
#define CFEAT 12
#define COUT  64
#define NKTOT 81920            // 2048*40
#define LTOT  (BATCH*NKTOT)    // 327680
#define SLOPEK 0.2f
#define EPSK   1e-5f
#define GROUPS 16              // 16-column groups per block (256 cols/block)

// ws layout (float offsets)
#define WS_M    0     // 12   : sum of y per feat channel
#define WS_S    16    // 144  : 12x12 second-moment of y
#define WS_SC0  160   // 64   : BN0 scale
#define WS_B0   224   // 64   : BN0 bias
#define WS_SUM1 288   // 64   : BN1 sum
#define WS_SQ1  352   // 64   : BN1 sumsq
#define WS_SC1  416   // 64
#define WS_B1   480   // 64
#define WS_ZEND 544
#define WS_W1H  1024  // _Float16 region: 24576 halfs (48KB), A-fragment packed

// ---------------- K0: zero stats + pack W1 into f16 A-fragment layout --------
// A-fragment (16-bit A 16x32, ISA 7.12.2): lane = M%16 ; element e (v=e/2,h=e%2):
// k_local = (v&3)*2 + h + 8*(lane/16) + 16*(v/4)
__global__ void k0_init_pack(const float* __restrict__ W1, float* __restrict__ ws) {
  int t = threadIdx.x;
  for (int j = t; j < WS_ZEND; j += 256) ws[j] = 0.f;
  _Float16* W1h = (_Float16*)(ws + WS_W1H);
  for (int idx = t; idx < 24576; idx += 256) {
    int e    = idx & 15;
    int lane = (idx >> 4) & 31;
    int kf   = (idx >> 9) & 1;
    int fi   = (idx >> 10) % 6;     // which x-channel slice of W1
    int rt   = idx / 6144;          // 16-row tile of out (0..3)
    int v = e >> 1, h = e & 1;
    int kloc = ((v & 3) << 1) + h + ((lane >> 4) << 3) + ((v >> 2) << 4);
    int k = kf * 32 + kloc;
    int row = (rt * 16 + (lane & 15)) * CIN + fi;   // W1 row = o*6 + i
    W1h[idx] = (_Float16)W1[row * COUT + k];
  }
}

// ---------------- K1: y first/second moments (for analytic BN0 stats) --------
__global__ void __launch_bounds__(256) k1_ymoments(const float* __restrict__ y,
                                                   float* __restrict__ ws) {
  __shared__ float ly[CFEAT][257];   // pad to kill bank conflicts
  int t = threadIdx.x;
  int bpb = NKTOT / 256;             // 320 blocks per batch
  int b  = blockIdx.x / bpb;
  int l0 = (blockIdx.x % bpb) * 256;
#pragma unroll
  for (int f = 0; f < CFEAT; ++f)
    ly[f][t] = y[((size_t)b * CFEAT + f) * NKTOT + l0 + t];
  __syncthreads();
  if (t < 12 + 144) {
    float acc = 0.f;
    if (t < 12) {
      for (int l = 0; l < 256; ++l) acc += ly[t][l];
      atomicAdd(&ws[WS_M + t], acc);
    } else {
      int q = t - 12, f = q / 12, f2 = q % 12;
      for (int l = 0; l < 256; ++l) acc += ly[f][l] * ly[f2][l];
      atomicAdd(&ws[WS_S + q], acc);
    }
  }
}

// ---------------- K2: BN0 scale/bias from y moments -------------------------
__global__ void k2_bn0(const float* __restrict__ W0, const float* __restrict__ g0,
                       const float* __restrict__ b0, float* __restrict__ ws) {
  int o = threadIdx.x;
  if (o >= COUT) return;
  const float invL = 1.0f / (float)LTOT;
  float w[CFEAT];
#pragma unroll
  for (int f = 0; f < CFEAT; ++f) w[f] = W0[o * CFEAT + f];
  float mean = 0.f;
#pragma unroll
  for (int f = 0; f < CFEAT; ++f) mean = fmaf(w[f], ws[WS_M + f], mean);
  mean *= invL;
  float ex2 = 0.f;
#pragma unroll
  for (int f = 0; f < CFEAT; ++f) {
    float tacc = 0.f;
#pragma unroll
    for (int f2 = 0; f2 < CFEAT; ++f2) tacc = fmaf(w[f2], ws[WS_S + f * 12 + f2], tacc);
    ex2 = fmaf(w[f], tacc, ex2);
  }
  ex2 *= invL;
  float var  = ex2 - mean * mean;
  float rstd = rsqrtf(var + EPSK);
  float sc   = g0[o] * rstd;
  ws[WS_SC0 + o] = sc;
  ws[WS_B0 + o]  = b0[o] - mean * sc;
}

// ---------------- K3: fused conv0+BN0+leaky -> WMMA GEMM -> x-contraction ----
// Block: 128 thr (4 waves), 256 columns per block (16 groups of 16), one batch.
// Wave rt owns out rows [16rt,16rt+16). A-fragments (W1 slices) live in VGPRs
// for the whole block; B (h') is produced per group into a double-buffered
// 2KB LDS tile -> one barrier per group.
__global__ void __launch_bounds__(128)
k3_main(const float* __restrict__ x, const float* __restrict__ y,
        const float* __restrict__ W0, const float* __restrict__ ws,
        const _Float16* __restrict__ W1h, float* __restrict__ out) {
  __shared__ __align__(32) _Float16 lB[2048];   // 4KB: double-buffered B tile
  __shared__ float lW0[COUT * CFEAT];
  __shared__ float lsc[COUT], lbs[COUT];

  int t  = threadIdx.x;
  int b  = blockIdx.y;
  int c0 = blockIdx.x * (GROUPS * 16);

  for (int j = t; j < COUT * CFEAT; j += 128) lW0[j] = W0[j];
  if (t < COUT) { lsc[t] = ws[WS_SC0 + t]; lbs[t] = ws[WS_B0 + t]; }

  int cl   = t & 15;          // column for h' producer
  int og0  = (t >> 4) << 3;   // 8-row strip of h' this thread produces
  int lane = t & 31;
  int wv   = t >> 5;          // rt: 16-row out tile
  int nlo  = lane & 15;       // wmma D column
  int hi   = lane >> 4;

  // ---- hoist A-fragments (W1 slices for this wave's row tile) into VGPRs ----
  // Coalesced: each lane reads its 32B fragment slice; W1h stays L2-resident.
  v16h A[2 * CIN];
  {
    const v16h* Ag = (const v16h*)W1h;
#pragma unroll
    for (int i = 0; i < CIN; ++i) {
      A[2 * i]     = Ag[(size_t)((wv * CIN + i) * 2 + 0) * 32 + lane];
      A[2 * i + 1] = Ag[(size_t)((wv * CIN + i) * 2 + 1) * 32 + lane];
    }
  }

  const float* ybase = y   + (size_t)b * CFEAT * NKTOT + c0;
  const float* xbase = x   + (size_t)b * CIN   * NKTOT + c0;
  float*       obase = out + (size_t)b * COUT  * NKTOT + c0;

  __syncthreads();   // lW0/lsc/lbs ready

  for (int g = 0; g < GROUPS; ++g) {
    int cg = g * 16;
    int buf = (g & 1) << 10;                           // half-index of lB buffer
    // prefetch next group's y columns into cache (global_prefetch_b8)
    if (g + 1 < GROUPS) {
#pragma unroll
      for (int f = 0; f < CFEAT; ++f)
        __builtin_prefetch(ybase + (size_t)f * NKTOT + cg + 16 + cl, 0, 0);
    }
    // ---- produce h' (64x16) directly into B-fragment layout ----
    float yv[CFEAT];
#pragma unroll
    for (int f = 0; f < CFEAT; ++f) yv[f] = ybase[(size_t)f * NKTOT + cg + cl];
#pragma unroll
    for (int oo = 0; oo < 8; ++oo) {
      int o = og0 + oo;
      float acc = 0.f;
#pragma unroll
      for (int f = 0; f < CFEAT; ++f) acc = fmaf(lW0[o * CFEAT + f], yv[f], acc);
      acc = fmaf(acc, lsc[o], lbs[o]);                 // BN0
      acc = acc > 0.f ? acc : SLOPEK * acc;            // leaky
      int kf = o >> 5, r = o & 31;
      int bl = (((r >> 3) & 1) << 4) + cl;             // fragment lane
      int e  = (r & 7) + ((r >> 4) << 3);              // element in v16h
      lB[buf + ((kf << 5) + bl) * 16 + e] = (_Float16)acc;
    }
    __syncthreads();   // single barrier per group (lB double-buffered)

    // ---- WMMA: u_i = W1_slice_i[16x64] @ h'[64x16] ----
    v16h bf0 = *((const v16h*)(lB + buf) + lane);
    v16h bf1 = *((const v16h*)(lB + buf) + 32 + lane);
    v8f u[CIN];
#pragma unroll
    for (int i = 0; i < CIN; ++i) {
      v8f acc = {0.f, 0.f, 0.f, 0.f, 0.f, 0.f, 0.f, 0.f};
      acc = __builtin_amdgcn_wmma_f32_16x16x32_f16(false, A[2 * i],     false, bf0,
                                                   (short)0, acc, false, false);
      acc = __builtin_amdgcn_wmma_f32_16x16x32_f16(false, A[2 * i + 1], false, bf1,
                                                   (short)0, acc, false, false);
      u[i] = acc;
    }

    // ---- fold in x: out[o,c] = sum_i x[i,c] * u_i[o,c] ----
    float xv[CIN];
#pragma unroll
    for (int i = 0; i < CIN; ++i) xv[i] = xbase[(size_t)i * NKTOT + cg + nlo];
#pragma unroll
    for (int j = 0; j < 8; ++j) {
      float r = xv[0] * u[0][j];
#pragma unroll
      for (int i = 1; i < CIN; ++i) r = fmaf(xv[i], u[i][j], r);
      int o = wv * 16 + j + (hi << 3);                 // D layout: M = j + 8*hi
      obase[(size_t)o * NKTOT + cg + nlo] = r;
    }
  }
}

// ---------------- K4: BN1 per-channel sum / sumsq over d_out -----------------
__global__ void __launch_bounds__(256) k4_stats(const float* __restrict__ out,
                                                float* __restrict__ ws) {
  __shared__ float ssum[8], ssq[8];
  int o = blockIdx.x, b = blockIdx.y;
  const float4* p = (const float4*)(out + ((size_t)b * COUT + o) * NKTOT);
  float s = 0.f, q = 0.f;
  for (int j = threadIdx.x; j < NKTOT / 4; j += 256) {
    float4 v = p[j];
    s += v.x + v.y + v.z + v.w;
    q += v.x * v.x + v.y * v.y + v.z * v.z + v.w * v.w;
  }
#pragma unroll
  for (int m = 16; m >= 1; m >>= 1) {
    s += __shfl_xor(s, m, 32);
    q += __shfl_xor(q, m, 32);
  }
  int lane = threadIdx.x & 31, wv = threadIdx.x >> 5;
  if (lane == 0) { ssum[wv] = s; ssq[wv] = q; }
  __syncthreads();
  if (threadIdx.x == 0) {
    float S = 0.f, Q = 0.f;
#pragma unroll
    for (int w = 0; w < 8; ++w) { S += ssum[w]; Q += ssq[w]; }
    atomicAdd(&ws[WS_SUM1 + o], S);
    atomicAdd(&ws[WS_SQ1  + o], Q);
  }
}

// ---------------- K5a: BN1 scale/bias ----------------------------------------
__global__ void k5a_bn1(const float* __restrict__ g1, const float* __restrict__ b1,
                        float* __restrict__ ws) {
  int o = threadIdx.x;
  if (o >= COUT) return;
  const float invL = 1.0f / (float)LTOT;
  float mean = ws[WS_SUM1 + o] * invL;
  float var  = ws[WS_SQ1 + o] * invL - mean * mean;
  float rstd = rsqrtf(var + EPSK);
  float sc   = g1[o] * rstd;
  ws[WS_SC1 + o] = sc;
  ws[WS_B1 + o]  = b1[o] - mean * sc;
}

// ---------------- K5b: apply BN1 + leaky in place ----------------------------
__global__ void __launch_bounds__(256) k5b_apply(float* __restrict__ out,
                                                 const float* __restrict__ ws) {
  size_t idx = (size_t)blockIdx.x * 256 + threadIdx.x;   // float4 index
  const size_t n4 = (size_t)LTOT * COUT / 4;
  if (idx >= n4) return;
  size_t e = idx * 4;
  int o = (int)((e / NKTOT) % COUT);                     // constant within NK run
  float sc = ws[WS_SC1 + o], bs = ws[WS_B1 + o];
  float4 v = ((float4*)out)[idx];
  v.x = fmaf(v.x, sc, bs); v.x = v.x > 0.f ? v.x : SLOPEK * v.x;
  v.y = fmaf(v.y, sc, bs); v.y = v.y > 0.f ? v.y : SLOPEK * v.y;
  v.z = fmaf(v.z, sc, bs); v.z = v.z > 0.f ? v.z : SLOPEK * v.z;
  v.w = fmaf(v.w, sc, bs); v.w = v.w > 0.f ? v.w : SLOPEK * v.w;
  ((float4*)out)[idx] = v;
}

extern "C" void kernel_launch(void* const* d_in, const int* in_sizes, int n_in,
                              void* d_out, int out_size, void* d_ws, size_t ws_size,
                              hipStream_t stream) {
  const float* x  = (const float*)d_in[0];
  const float* y  = (const float*)d_in[1];
  const float* W0 = (const float*)d_in[2];
  const float* g0 = (const float*)d_in[3];
  const float* b0 = (const float*)d_in[4];
  const float* W1 = (const float*)d_in[5];
  const float* g1 = (const float*)d_in[6];
  const float* b1 = (const float*)d_in[7];
  float* out = (float*)d_out;
  float* ws  = (float*)d_ws;
  const _Float16* W1h = (const _Float16*)(ws + WS_W1H);

  k0_init_pack<<<1, 256, 0, stream>>>(W1, ws);
  k1_ymoments<<<LTOT / 256, 256, 0, stream>>>(y, ws);
  k2_bn0<<<1, 64, 0, stream>>>(W0, g0, b0, ws);
  dim3 g3(NKTOT / (GROUPS * 16), BATCH);
  k3_main<<<g3, 128, 0, stream>>>(x, y, W0, ws, W1h, out);
  dim3 g4(COUT, BATCH);
  k4_stats<<<g4, 256, 0, stream>>>(out, ws);
  k5a_bn1<<<1, 64, 0, stream>>>(g1, b1, ws);
  k5b_apply<<<(LTOT * COUT / 4 + 255) / 256, 256, 0, stream>>>(out, ws);
}
